// Attention_83476984365503
// MI455X (gfx1250) — matrix operand order repaired and checked by reading, not compiled, not run on hardware
//
#include <hip/hip_runtime.h>
#include <hip/hip_bf16.h>

typedef __attribute__((ext_vector_type(16))) __bf16 v16bf;
typedef __attribute__((ext_vector_type(8)))  float  v8f;

#define HID 1024
#define SPC 2048         // spatial channels C
#define NB  256          // batch
#define NSP 196          // H*W
#define KT  (SPC / 32)   // 64 K-tiles of 32
#define DT  (HID / 16)   // 64 d-tiles of 16
#define MROWS 32         // rows (n) per workgroup
#define KCH 512          // K columns staged per LDS chunk
#define LDA (KCH + 8)    // padded LDS row stride (ushorts)

__device__ __forceinline__ unsigned short f32_to_bf16(float f) {
  union { float f; unsigned u; } v; v.f = f;
  unsigned r = v.u + 0x7FFFu + ((v.u >> 16) & 1u);   // round-to-nearest-even
  return (unsigned short)(r >> 16);
}

// ---------------------------------------------------------------------------
// Pack W1s (rows 1024..3071 of W1) into bf16 WMMA B-fragment layout:
// frag index = ((kt*DT + dt)*32 + lane), 16 contiguous ushorts per lane.
// B order used here (32x16, bf16, contiguous k, same as A): lane<16 -> N=lane, K=0..15 ; lane>=16 -> N=lane-16, K=16..31
// ---------------------------------------------------------------------------
__global__ void pack_w1s_kernel(const float* __restrict__ W1,
                                unsigned short* __restrict__ out) {
  int t = blockIdx.x * blockDim.x + threadIdx.x;   // KT*DT*32 threads
  int lane = t & 31;
  int dt   = (t >> 5) & (DT - 1);
  int kt   = t >> 11;
  if (kt >= KT) return;
  int col  = lane & 15;
  int ksel = (lane >> 4) << 4;                     // 0 or 16
  const float* w1s = W1 + (size_t)HID * HID;       // skip W1h block
  unsigned short frag[16];
#pragma unroll
  for (int e = 0; e < 16; ++e) {
    int k = kt * 32 + ksel + e;
    int d = dt * 16 + col;
    frag[e] = f32_to_bf16(w1s[(size_t)k * HID + d]);
  }
  unsigned short* dst = out + ((((size_t)kt * DT + dt) * 32 + lane) << 4);
  *(uint4*)(dst)     = *(const uint4*)(frag);
  *(uint4*)(dst + 8) = *(const uint4*)(frag + 8);
}

// ---------------------------------------------------------------------------
// hpre[b][d] = hidden[b] @ W1h[:,d] + b1[d]
// 16 batches per block: each W1h element loaded once feeds 16 FMAs; hidden
// accesses are wave-uniform -> scalar loads through the constant cache.
// ---------------------------------------------------------------------------
__global__ void __launch_bounds__(256)
hpre_kernel(const float* __restrict__ hidden,
            const float* __restrict__ W1,
            const float* __restrict__ b1,
            float* __restrict__ hpre) {
  int d  = blockIdx.x * blockDim.x + threadIdx.x;  // 0..1023
  int b0 = blockIdx.y * 16;
  float acc[16];
#pragma unroll
  for (int j = 0; j < 16; ++j) acc[j] = 0.0f;
  for (int k = 0; k < HID; ++k) {
    float w = W1[(size_t)k * HID + d];
#pragma unroll
    for (int j = 0; j < 16; ++j)
      acc[j] = fmaf(hidden[(size_t)(b0 + j) * HID + k], w, acc[j]);
  }
  float bb = b1[d];
#pragma unroll
  for (int j = 0; j < 16; ++j)
    hpre[(size_t)(b0 + j) * HID + d] = acc[j] + bb;
}

// ---------------------------------------------------------------------------
// Fused score GEMM: per (b, 32-row n-block) compute
//   scores[n] = sum_d tanh( (sp[b] @ W1s)[n,d] + hpre[b,d] ) * W2[d]
// 8 waves x (2 row-tiles x 8 d-tiles) accumulators; each B fragment is loaded
// once and consumed by two WMMAs. B fragments use a 4-deep register pipeline
// (4 simultaneously-live buffers) so 6-8 global_load_b128 stay in flight.
// ---------------------------------------------------------------------------
__global__ void __launch_bounds__(256)
attn_scores_kernel(const float* __restrict__ spatial,
                   const unsigned short* __restrict__ wpack,
                   const float* __restrict__ hpre,
                   const float* __restrict__ W2,
                   float* __restrict__ scores) {
  __shared__ unsigned short Ash[MROWS * LDA];   // 32 x 512 bf16 A chunk (~33 KB)
  __shared__ float scoreSh[MROWS];

  const int tid  = threadIdx.x;
  const int lane = tid & 31;
  const int wave = tid >> 5;
  const int n0   = blockIdx.x * MROWS;          // 0,32,...,192
  const int b    = blockIdx.y;

  const float* spb = spatial + (size_t)b * SPC * NSP;

  if (tid < MROWS) scoreSh[tid] = 0.0f;

  v8f acc[16] = {};                             // [t*2 + r] : 16 tiles of 16x16 f32
  const int mrow = lane & 15;
  const int asel = (lane >> 4) << 4;            // A layout: 0 or 16

  for (int kc = 0; kc < SPC / KCH; ++kc) {      // four 512-wide K chunks
    __syncthreads();                            // prior compute done with Ash
    // stage: sp element (n, c) lives at spb[c*NSP + n] (n contiguous -> coalesced)
#pragma unroll 4
    for (int i = 0; i < (MROWS * KCH) / 256; ++i) {
      int idx = i * 256 + tid;
      int n   = idx & (MROWS - 1);
      int cl  = idx >> 5;                       // 0..511
      int nn  = n0 + n;
      float v = (nn < NSP) ? spb[((size_t)(kc * KCH + cl)) * NSP + nn] : 0.0f;
      Ash[n * LDA + cl] = f32_to_bf16(v);
    }
    __syncthreads();

    for (int kt2 = 0; kt2 < KCH / 32; ++kt2) {
      const int kt = kc * (KCH / 32) + kt2;
      // A fragments for both row-tiles (16-bit A 16x32 layout)
      v16bf afrag[2];
#pragma unroll
      for (int r = 0; r < 2; ++r) {
        const unsigned short* ap = &Ash[(r * 16 + mrow) * LDA + kt2 * 32 + asel];
        union { uint4 u[2]; v16bf v; } ac;
        ac.u[0] = *(const uint4*)(ap);
        ac.u[1] = *(const uint4*)(ap + 8);
        afrag[r] = ac.v;
      }

      const unsigned short* bbase =
          wpack + ((((size_t)kt * DT + wave * 8) * 32 + lane) << 4);

      // 4-deep B-fragment pipeline: all four buffers are live at once so the
      // register allocator cannot collapse them; slot t&3 is consumed by two
      // WMMAs then refilled with fragment t+4.
      union BB { uint4 u[2]; v16bf v; };
      BB bc[4];
#pragma unroll
      for (int p = 0; p < 4; ++p) {
        const unsigned short* bp = bbase + ((size_t)p << 9);
        bc[p].u[0] = *(const uint4*)(bp);
        bc[p].u[1] = *(const uint4*)(bp + 8);
      }
#pragma unroll
      for (int t = 0; t < 8; ++t) {
        const int slot = t & 3;
        acc[t * 2 + 0] = __builtin_amdgcn_wmma_f32_16x16x32_bf16(
            false, afrag[0], false, bc[slot].v, (short)0, acc[t * 2 + 0], false, false);
        acc[t * 2 + 1] = __builtin_amdgcn_wmma_f32_16x16x32_bf16(
            false, afrag[1], false, bc[slot].v, (short)0, acc[t * 2 + 1], false, false);
        if (t < 4) {
          const unsigned short* bp = bbase + ((size_t)(t + 4) << 9);
          bc[slot].u[0] = *(const uint4*)(bp);
          bc[slot].u[1] = *(const uint4*)(bp + 8);
        }
      }
    }
  }

  // Epilogue: C/D layout -> VGPR v, lanes<16: row=v, N=lane ; lanes>=16: row=v+8
  const float* hpb  = hpre + (size_t)b * HID;
  const int col   = lane & 15;
  const int rbase = (lane >> 4) << 3;
  float part[2][8];
#pragma unroll
  for (int r = 0; r < 2; ++r)
#pragma unroll
    for (int v = 0; v < 8; ++v) part[r][v] = 0.0f;
#pragma unroll
  for (int t = 0; t < 8; ++t) {
    int d = wave * 128 + t * 16 + col;
    float hp = hpb[d];
    float w2 = W2[d];
#pragma unroll
    for (int r = 0; r < 2; ++r)
#pragma unroll
      for (int v = 0; v < 8; ++v)
        part[r][v] += tanhf(acc[t * 2 + r][v] + hp) * w2;
  }
#pragma unroll
  for (int r = 0; r < 2; ++r)
#pragma unroll
    for (int v = 0; v < 8; ++v)
      atomicAdd(&scoreSh[r * 16 + rbase + v], part[r][v]);
  __syncthreads();

  if (tid < MROWS && (n0 + tid) < NSP)
    scores[(size_t)b * 256 + n0 + tid] = scoreSh[tid];
}

// ---------------------------------------------------------------------------
// softmax over n=196 per batch (b2 is shift-invariant -> omitted)
// ---------------------------------------------------------------------------
__global__ void softmax_kernel(const float* __restrict__ scores,
                               float* __restrict__ attn) {
  __shared__ float red[256];
  int b = blockIdx.x, tid = threadIdx.x;
  float s = (tid < NSP) ? scores[(size_t)b * 256 + tid] : -1e30f;
  red[tid] = s;
  __syncthreads();
  for (int off = 128; off > 0; off >>= 1) {
    if (tid < off) red[tid] = fmaxf(red[tid], red[tid + off]);
    __syncthreads();
  }
  float m = red[0];
  __syncthreads();
  float e = (tid < NSP) ? __expf(s - m) : 0.0f;
  red[tid] = e;
  __syncthreads();
  for (int off = 128; off > 0; off >>= 1) {
    if (tid < off) red[tid] += red[tid + off];
    __syncthreads();
  }
  float inv = 1.0f / red[0];
  if (tid < NSP) attn[(size_t)b * NSP + tid] = e * inv;
}

// ---------------------------------------------------------------------------
// context[b][c] = sum_n attn[b][n] * spatial[b][c][n]  (float4 over contiguous n)
// ---------------------------------------------------------------------------
__global__ void context_kernel(const float* __restrict__ spatial,
                               const float* __restrict__ attn,
                               float* __restrict__ context) {
  __shared__ float a[NSP];
  int b = blockIdx.y, tid = threadIdx.x;
  if (tid < NSP) a[tid] = attn[(size_t)b * NSP + tid];
  __syncthreads();
  int c = blockIdx.x * 256 + tid;
  const float4* p = (const float4*)(spatial + ((size_t)b * SPC + c) * NSP);
  float sum = 0.0f;
#pragma unroll 7
  for (int i = 0; i < NSP / 4; ++i) {
    float4 v = p[i];
    sum += a[4 * i] * v.x + a[4 * i + 1] * v.y + a[4 * i + 2] * v.z + a[4 * i + 3] * v.w;
  }
  context[(size_t)b * SPC + c] = sum;
}

// ---------------------------------------------------------------------------
extern "C" void kernel_launch(void* const* d_in, const int* in_sizes, int n_in,
                              void* d_out, int out_size, void* d_ws, size_t ws_size,
                              hipStream_t stream) {
  (void)in_sizes; (void)n_in; (void)out_size; (void)ws_size;
  const float* hidden  = (const float*)d_in[0];
  const float* spatial = (const float*)d_in[1];
  const float* W1      = (const float*)d_in[2];
  const float* b1      = (const float*)d_in[3];
  const float* W2      = (const float*)d_in[4];
  // d_in[5] = b2: softmax is shift-invariant, not needed for attn or context.

  float* out     = (float*)d_out;
  float* context = out;                        // 256*2048 floats
  float* attn    = out + (size_t)NB * SPC;     // 256*196 floats

  // workspace: [0,4MB) packed bf16 W1s | [4MB,5MB) hpre | [5MB,+256KB) scores
  unsigned short* wpack = (unsigned short*)d_ws;
  float* hpre   = (float*)((char*)d_ws + (size_t)4 * 1024 * 1024);
  float* scores = (float*)((char*)d_ws + (size_t)5 * 1024 * 1024);

  pack_w1s_kernel<<<dim3((KT * DT * 32) / 256), dim3(256), 0, stream>>>(W1, wpack);
  hpre_kernel<<<dim3(HID / 256, NB / 16), dim3(256), 0, stream>>>(hidden, W1, b1, hpre);
  attn_scores_kernel<<<dim3(7, NB), dim3(256), 0, stream>>>(spatial, wpack, hpre, W2, scores);
  softmax_kernel<<<dim3(NB), dim3(256), 0, stream>>>(scores, attn);
  context_kernel<<<dim3(SPC / 256, NB), dim3(256), 0, stream>>>(spatial, attn, context);
}
